// AtomAttentionEncoder_85899345920320
// MI455X (gfx1250) — matrix operand order repaired and checked by reading, not compiled, not run on hardware
//
#include <hip/hip_runtime.h>
#include <hip/hip_bf16.h>

typedef __attribute__((ext_vector_type(16))) _Float16 v16h;
typedef __attribute__((ext_vector_type(8)))  _Float16 v8h;
typedef __attribute__((ext_vector_type(8)))  float    v8f;

#define NATOMS 1152
#define CAE    128      // atom embedding
#define CPE    16       // pair embedding
#define CTE    384      // token embedding
#define NBLK   3
#define NHEAD  4
#define DHEAD  32
#define NQW    32       // query window
#define NKW    128      // key window
#define NEGBIG (-1.0e9f)

#define EP_NONE       0
#define EP_SIGMOID    1
#define EP_SIGMOID_M2 2
#define EP_RELU       3
#define EP_GATED_ACC  4
#define EP_SILU       5
#define EP_MUL        6

__device__ __forceinline__ float sigm(float x) { return 1.0f / (1.0f + __expf(-x)); }
__device__ __forceinline__ int   iclamp(int v, int lo, int hi) { return v < lo ? lo : (v > hi ? hi : v); }

// ---------------------------------------------------------------------------
// weight prep: Wt[n][k] = (f16) W[k][n]  (transposed f16 so the WMMA B-fragment
// is a contiguous 32B load per lane)
// ---------------------------------------------------------------------------
__global__ void convert_wt(const float* __restrict__ W, _Float16* __restrict__ Wt,
                           int K, int Nc)
{
    int idx = blockIdx.x * blockDim.x + threadIdx.x;
    int n = idx / K, k = idx % K;
    Wt[idx] = (_Float16)W[(size_t)k * Nc + n];
}

// ---------------------------------------------------------------------------
// c = pos @ W_pos ; sn = LN(c) ; a = c ; cc = relu(c) @ W_cond
// ---------------------------------------------------------------------------
__global__ void init_kernel(const float* __restrict__ pos, const float* __restrict__ W_pos,
                            const float* __restrict__ W_cond,
                            float* __restrict__ c_out, float* __restrict__ sn,
                            float* __restrict__ a, float* __restrict__ cc)
{
    int i = blockIdx.x, t = threadIdx.x;
    float p0 = pos[i * 3 + 0], p1 = pos[i * 3 + 1], p2 = pos[i * 3 + 2];
    float cv = p0 * W_pos[t] + p1 * W_pos[CAE + t] + p2 * W_pos[2 * CAE + t];
    c_out[i * CAE + t] = cv;
    a[i * CAE + t] = cv;

    __shared__ float s1[CAE], s2[CAE], rc[CAE];
    s1[t] = cv; s2[t] = cv * cv; rc[t] = fmaxf(cv, 0.0f);
    __syncthreads();
    for (int off = 64; off > 0; off >>= 1) {
        if (t < off) { s1[t] += s1[t + off]; s2[t] += s2[t + off]; }
        __syncthreads();
    }
    float mu = s1[0] * (1.0f / CAE);
    float var = s2[0] * (1.0f / CAE) - mu * mu;
    float rs = rsqrtf(var + 1e-5f);
    sn[i * CAE + t] = (cv - mu) * rs;

    if (t < CPE) {
        float acc = 0.0f;
        for (int k = 0; k < CAE; ++k) acc += rc[k] * W_cond[k * CPE + t];
        cc[i * CPE + t] = acc;
    }
}

// ---------------------------------------------------------------------------
// pr[i,j,:] full pair tensor + 3-layer 16x16 MLP residual, one thread per pair
// ---------------------------------------------------------------------------
__global__ void pair_kernel(const float* __restrict__ pos, const int* __restrict__ uid,
                            const float* __restrict__ cc,
                            const float* __restrict__ W_off, const float* __restrict__ W_inv,
                            const float* __restrict__ W_mask,
                            const float* __restrict__ W_p1, const float* __restrict__ W_p2,
                            const float* __restrict__ W_p3,
                            float* __restrict__ pr)
{
    __shared__ float sW1[CPE * CPE], sW2[CPE * CPE], sW3[CPE * CPE];
    __shared__ float sOff[3 * CPE], sInv[CPE], sMsk[CPE];
    int t = threadIdx.x;
    if (t < CPE * CPE) { sW1[t] = W_p1[t]; sW2[t] = W_p2[t]; sW3[t] = W_p3[t]; }
    if (t < 3 * CPE) sOff[t] = W_off[t];
    if (t < CPE) { sInv[t] = W_inv[t]; sMsk[t] = W_mask[t]; }
    __syncthreads();

    int idx = blockIdx.x * blockDim.x + t;
    int i = idx / NATOMS, j = idx % NATOMS;
    float d0 = pos[i * 3 + 0] - pos[j * 3 + 0];
    float d1 = pos[i * 3 + 1] - pos[j * 3 + 1];
    float d2 = pos[i * 3 + 2] - pos[j * 3 + 2];
    float m = (uid[i] == uid[j]) ? 1.0f : 0.0f;
    float inv = 1.0f / (1.0f + d0 * d0 + d1 * d1 + d2 * d2);

    float v[CPE], h1[CPE], h2[CPE], h3[CPE];
#pragma unroll
    for (int c = 0; c < CPE; ++c)
        v[c] = m * (d0 * sOff[c] + d1 * sOff[CPE + c] + d2 * sOff[2 * CPE + c]
                    + inv * sInv[c] + sMsk[c])
               + cc[i * CPE + c] + cc[j * CPE + c];
#pragma unroll
    for (int o = 0; o < CPE; ++o) {
        float acc = 0.0f;
#pragma unroll
        for (int c = 0; c < CPE; ++c) acc += fmaxf(v[c], 0.0f) * sW1[c * CPE + o];
        h1[o] = acc;
    }
#pragma unroll
    for (int o = 0; o < CPE; ++o) {
        float acc = 0.0f;
#pragma unroll
        for (int c = 0; c < CPE; ++c) acc += fmaxf(h1[c], 0.0f) * sW2[c * CPE + o];
        h2[o] = acc;
    }
#pragma unroll
    for (int o = 0; o < CPE; ++o) {
        float acc = 0.0f;
#pragma unroll
        for (int c = 0; c < CPE; ++c) acc += fmaxf(h2[c], 0.0f) * sW3[c * CPE + o];
        h3[o] = acc;
    }
    size_t base = ((size_t)i * NATOMS + j) * CPE;
#pragma unroll
    for (int c = 0; c < CPE; ++c) pr[base + c] = v[c] + h3[c];
}

// ---------------------------------------------------------------------------
// local-window pair bias for all 3 blocks x 4 heads; OOR keys -> -1e9
// ---------------------------------------------------------------------------
__global__ void bias_kernel(const float* __restrict__ pr,
                            const float* __restrict__ Wz0, const float* __restrict__ Wz1,
                            const float* __restrict__ Wz2,
                            float* __restrict__ biasws)
{
    const float* Wzs[NBLK] = { Wz0, Wz1, Wz2 };
    int gid = blockIdx.x * blockDim.x + threadIdx.x;   // over NATOMS*NKW
    int qrow = gid / NKW, kl = gid % NKW;
    int kidx = (qrow / NQW) * NQW - 48 + kl;
    bool valid = (kidx >= 0) && (kidx < NATOMS);

    float z[CPE];
    if (valid) {
        size_t base = ((size_t)qrow * NATOMS + kidx) * CPE;
        float mu = 0.0f;
#pragma unroll
        for (int c = 0; c < CPE; ++c) { z[c] = pr[base + c]; mu += z[c]; }
        mu *= (1.0f / CPE);
        float var = 0.0f;
#pragma unroll
        for (int c = 0; c < CPE; ++c) { float d = z[c] - mu; var += d * d; }
        var *= (1.0f / CPE);
        float rs = rsqrtf(var + 1e-5f);
#pragma unroll
        for (int c = 0; c < CPE; ++c) z[c] = (z[c] - mu) * rs;
    }
    for (int b = 0; b < NBLK; ++b) {
        for (int h = 0; h < NHEAD; ++h) {
            float val = NEGBIG;
            if (valid) {
                float acc = 0.0f;
#pragma unroll
                for (int c = 0; c < CPE; ++c) acc += z[c] * Wzs[b][c * NHEAD + h];
                val = acc;
            }
            biasws[(((size_t)b * NHEAD + h) * NATOMS + qrow) * NKW + kl] = val;
        }
    }
}

// ---------------------------------------------------------------------------
// WMMA GEMM, compile-time-specialized epilogue.
//   C = epilogue(A[M,K] @ Wt^T (+bias))  with Wt stored f16 transposed [Nc][K].
// blockDim 256 (8 waves), wave -> one 16x16 C tile, f16 WMMA K-step 32.
// A fragment: 4x b128 loads; B fragment: 1x contiguous v16h load.
// ---------------------------------------------------------------------------
template <int MODE, bool HASBIAS, bool OUTH>
__global__ void wmma_gemm(const float* __restrict__ A, const _Float16* __restrict__ Wt,
                          const float* __restrict__ bias, const float* __restrict__ gate,
                          float* __restrict__ Cf, _Float16* __restrict__ Ch,
                          int M, int K, int Nc)
{
    int wave = threadIdx.x >> 5;
    int lane = threadIdx.x & 31;
    int tm = blockIdx.x;
    int tn = blockIdx.y * 8 + wave;
    if (tn * 16 >= Nc) return;                 // wave-uniform
    int ln16 = lane & 15;
    int hi = lane >> 4;
    int mrow = tm * 16 + ln16;
    int ncol = tn * 16 + ln16;

    v8f acc = {};
    const float*    Ar = A  + (size_t)mrow * K;
    const _Float16* Wr = Wt + (size_t)ncol * K;
    for (int k0 = 0; k0 < K; k0 += 32) {
        int ka = k0 + hi * 16;
        float4 f0 = *(const float4*)(Ar + ka);
        float4 f1 = *(const float4*)(Ar + ka + 4);
        float4 f2 = *(const float4*)(Ar + ka + 8);
        float4 f3 = *(const float4*)(Ar + ka + 12);
        v16h af;
        af[0]  = (_Float16)f0.x; af[1]  = (_Float16)f0.y;
        af[2]  = (_Float16)f0.z; af[3]  = (_Float16)f0.w;
        af[4]  = (_Float16)f1.x; af[5]  = (_Float16)f1.y;
        af[6]  = (_Float16)f1.z; af[7]  = (_Float16)f1.w;
        af[8]  = (_Float16)f2.x; af[9]  = (_Float16)f2.y;
        af[10] = (_Float16)f2.z; af[11] = (_Float16)f2.w;
        af[12] = (_Float16)f3.x; af[13] = (_Float16)f3.y;
        af[14] = (_Float16)f3.z; af[15] = (_Float16)f3.w;
        v16h bf = *(const v16h*)(Wr + k0 + hi * 16);
        acc = __builtin_amdgcn_wmma_f32_16x16x32_f16(false, af, false, bf,
                                                     (short)0, acc, false, false);
    }
#pragma unroll
    for (int r = 0; r < 8; ++r) {
        int row = tm * 16 + hi * 8 + r;
        size_t idx = (size_t)row * Nc + ncol;
        float v = acc[r];
        if constexpr (HASBIAS) v += bias[ncol];
        if constexpr (OUTH) {
            Ch[idx] = (_Float16)v;                       // MODE == EP_NONE path
        } else if constexpr (MODE == EP_SIGMOID) {
            Cf[idx] = sigm(v);
        } else if constexpr (MODE == EP_SIGMOID_M2) {
            Cf[idx] = sigm(v - 2.0f);
        } else if constexpr (MODE == EP_RELU) {
            Cf[idx] = fmaxf(v, 0.0f);
        } else if constexpr (MODE == EP_SILU) {
            Cf[idx] = v * sigm(v);
        } else if constexpr (MODE == EP_GATED_ACC) {
            Cf[idx] += gate[idx] * v;
        } else if constexpr (MODE == EP_MUL) {
            Cf[idx] = gate[idx] * v;
        } else {
            Cf[idx] = v;
        }
    }
}

// ---------------------------------------------------------------------------
// adaLN: out = sig * LN(x) + add   (one block per row, 128 threads)
// ---------------------------------------------------------------------------
__global__ void adaln_kernel(const float* __restrict__ x, const float* __restrict__ sig,
                             const float* __restrict__ add, float* __restrict__ out)
{
    int row = blockIdx.x, t = threadIdx.x;
    float v = x[row * CAE + t];
    __shared__ float s1[CAE], s2[CAE];
    s1[t] = v; s2[t] = v * v;
    __syncthreads();
    for (int off = 64; off > 0; off >>= 1) {
        if (t < off) { s1[t] += s1[t + off]; s2[t] += s2[t + off]; }
        __syncthreads();
    }
    float mu = s1[0] * (1.0f / CAE);
    float var = s2[0] * (1.0f / CAE) - mu * mu;
    float rs = rsqrtf(var + 1e-5f);
    int idx = row * CAE + t;
    out[idx] = sig[idx] * ((v - mu) * rs) + add[idx];
}

// ---------------------------------------------------------------------------
// windowed attention on f16 Q/K/V: grid (N/NQW, NHEAD), 128 threads (4 waves).
// logits 32x128 via 16 WMMAs, f32 softmax in LDS, P@V via 16 WMMAs.
// ---------------------------------------------------------------------------
__global__ void attn_kernel(const _Float16* __restrict__ qh, const _Float16* __restrict__ kh,
                            const _Float16* __restrict__ vh, const float* __restrict__ biasb,
                            const float* __restrict__ gg, float* __restrict__ og)
{
    int qb = blockIdx.x, h = blockIdx.y;
    int wave = threadIdx.x >> 5, lane = threadIdx.x & 31;
    int ln16 = lane & 15, hi = lane >> 4;
    int kwin = qb * NQW - 48;
    const float scale = 0.17677669529663687f;   // 1/sqrt(32)

    __shared__ float P[NQW][NKW];               // 16 KB

    // phase 1: logits = Q . K^T * scale + bias
    for (int tile = wave; tile < 16; tile += 4) {
        int tmq = tile >> 3;        // 0..1 query-row tile
        int tnk = tile & 7;         // 0..7 key tile
        int mrow = qb * NQW + tmq * 16 + ln16;
        const _Float16* Ar = qh + (size_t)mrow * CAE + h * DHEAD;
        int ka = hi * 16;
        v8h a0 = *(const v8h*)(Ar + ka);
        v8h a1 = *(const v8h*)(Ar + ka + 8);
        v16h af = __builtin_shufflevector(a0, a1, 0, 1, 2, 3, 4, 5, 6, 7,
                                                  8, 9, 10, 11, 12, 13, 14, 15);
        int nk = iclamp(kwin + tnk * 16 + ln16, 0, NATOMS - 1);
        v16h bf = *(const v16h*)(kh + (size_t)nk * CAE + h * DHEAD + hi * 16);
        v8f acc = {};
        acc = __builtin_amdgcn_wmma_f32_16x16x32_f16(false, af, false, bf,
                                                     (short)0, acc, false, false);
#pragma unroll
        for (int r = 0; r < 8; ++r) {
            int rq = tmq * 16 + hi * 8 + r;
            int ck = tnk * 16 + ln16;
            P[rq][ck] = acc[r] * scale
                      + biasb[((size_t)h * NATOMS + qb * NQW + rq) * NKW + ck];
        }
    }
    __syncthreads();

    // phase 2: row softmax (thread per query row)
    if (threadIdx.x < NQW) {
        int rq = threadIdx.x;
        float mx = -3.4e38f;
        for (int k = 0; k < NKW; ++k) mx = fmaxf(mx, P[rq][k]);
        float s = 0.0f;
        for (int k = 0; k < NKW; ++k) { float e = __expf(P[rq][k] - mx); P[rq][k] = e; s += e; }
        float inv = 1.0f / s;
        for (int k = 0; k < NKW; ++k) P[rq][k] *= inv;
    }
    __syncthreads();

    // phase 3: O = P @ V (wave -> one 16x16 tile of the 32x32 head output)
    int tmq = wave >> 1, tdh = wave & 1;
    v8f acc = {};
    for (int k0 = 0; k0 < NKW; k0 += 32) {
        int arow = tmq * 16 + ln16;
        int ka = k0 + hi * 16;
        float4 p0 = *(const float4*)(&P[arow][ka]);
        float4 p1 = *(const float4*)(&P[arow][ka + 4]);
        float4 p2 = *(const float4*)(&P[arow][ka + 8]);
        float4 p3 = *(const float4*)(&P[arow][ka + 12]);
        v16h af;
        af[0]  = (_Float16)p0.x; af[1]  = (_Float16)p0.y;
        af[2]  = (_Float16)p0.z; af[3]  = (_Float16)p0.w;
        af[4]  = (_Float16)p1.x; af[5]  = (_Float16)p1.y;
        af[6]  = (_Float16)p1.z; af[7]  = (_Float16)p1.w;
        af[8]  = (_Float16)p2.x; af[9]  = (_Float16)p2.y;
        af[10] = (_Float16)p2.z; af[11] = (_Float16)p2.w;
        af[12] = (_Float16)p3.x; af[13] = (_Float16)p3.y;
        af[14] = (_Float16)p3.z; af[15] = (_Float16)p3.w;
        int ncol = tdh * 16 + ln16;
        int kb = k0 + hi * 16;
        v16h bf;
#pragma unroll
        for (int e = 0; e < 16; ++e) {
            int key = iclamp(kwin + kb + e, 0, NATOMS - 1);
            bf[e] = vh[(size_t)key * CAE + h * DHEAD + ncol];
        }
        acc = __builtin_amdgcn_wmma_f32_16x16x32_f16(false, af, false, bf,
                                                     (short)0, acc, false, false);
    }
#pragma unroll
    for (int r = 0; r < 8; ++r) {
        int row = qb * NQW + tmq * 16 + hi * 8 + r;
        int col = h * DHEAD + tdh * 16 + ln16;
        size_t idx = (size_t)row * CAE + col;
        og[idx] = acc[r] * gg[idx];
    }
}

// ---------------------------------------------------------------------------
// residue pooling: a_out[r, t] = mean over 3 atoms
// ---------------------------------------------------------------------------
__global__ void pool_kernel(const float* __restrict__ qp, float* __restrict__ aout)
{
    int idx = blockIdx.x * blockDim.x + threadIdx.x;   // over 384*CTE
    int r = idx / CTE, ch = idx % CTE;
    float s = qp[(size_t)(r * 3 + 0) * CTE + ch]
            + qp[(size_t)(r * 3 + 1) * CTE + ch]
            + qp[(size_t)(r * 3 + 2) * CTE + ch];
    aout[idx] = s * (1.0f / 3.0f);
}

// ---------------------------------------------------------------------------
extern "C" void kernel_launch(void* const* d_in, const int* in_sizes, int n_in,
                              void* d_out, int out_size, void* d_ws, size_t ws_size,
                              hipStream_t stream)
{
    (void)in_sizes; (void)n_in; (void)out_size; (void)ws_size;

    const float* ref_pos = (const float*)d_in[0];
    const int*   uid     = (const int*)  d_in[1];
    const float* W_pos   = (const float*)d_in[2];
    const float* W_off   = (const float*)d_in[3];
    const float* W_inv   = (const float*)d_in[4];
    const float* W_maskw = (const float*)d_in[5];
    const float* W_cond  = (const float*)d_in[6];
    const float* W_p1    = (const float*)d_in[7];
    const float* W_p2    = (const float*)d_in[8];
    const float* W_p3    = (const float*)d_in[9];
    const float* W_proj  = (const float*)d_in[10];
    // per-block params in insertion order:
    // 0 Wz 1 Wag 2 bag 3 Wab 4 Wq 5 bq 6 Wk 7 Wv 8 Wg 9 Wo 10 Wso
    // 11 Wtg 12 btg 13 Wtb 14 Wt1 15 Wt2 16 Wt3 17 Wts
    auto BP = [&](int b, int j) -> const float* {
        return (const float*)d_in[11 + b * 18 + j];
    };

    float* out   = (float*)d_out;
    float* outA  = out;                       // [384, 384]
    float* outQ  = outA + 384 * CTE;          // [1152, 384]
    float* outC  = outQ + NATOMS * CTE;       // [1152, 128]
    float* outPR = outC + NATOMS * CAE;       // [1152, 1152, 16]

    float* w = (float*)d_ws;
    float* sn     = w; w += NATOMS * CAE;
    float* aact   = w; w += NATOMS * CAE;
    float* cc     = w; w += NATOMS * CPE;
    float* ai     = w; w += NATOMS * CAE;
    float* sig_g  = w; w += NATOMS * CAE;
    float* b_ab   = w; w += NATOMS * CAE;
    float* gate_o = w; w += NATOMS * CAE;
    float* sig_t  = w; w += NATOMS * CAE;
    float* b_tb   = w; w += NATOMS * CAE;
    float* gate_t = w; w += NATOMS * CAE;
    float* gbuf   = w; w += NATOMS * CAE;
    float* obuf   = w; w += NATOMS * CAE;
    float* ti     = w; w += NATOMS * CAE;
    float* t1     = w; w += NATOMS * 2 * CAE;
    float* tsw    = w; w += NATOMS * 2 * CAE;
    float* biasws = w; w += (size_t)NBLK * NHEAD * NATOMS * NKW;

    // f16 region (32B-aligned: all prior float counts are multiples of 8)
    _Float16* hw = (_Float16*)w;
    size_t hoff = 0;
    auto halloc = [&](size_t n) { _Float16* p = hw + hoff; hoff += n; return p; };
    _Float16* qh = halloc(NATOMS * CAE);
    _Float16* kh = halloc(NATOMS * CAE);
    _Float16* vh = halloc(NATOMS * CAE);
    // transposed f16 weights, per block:
    // j: 0 Wag 1 Wab 2 Wq 3 Wk 4 Wv 5 Wg 6 Wo 7 Wso 8 Wtg 9 Wtb 10 Wts
    //    11 Wt1 12 Wt2 13 Wt3
    const int wsel[14]  = { 1, 3, 4, 6, 7, 8, 9, 10, 11, 13, 17, 14, 15, 16 };
    const int wK[14]    = { CAE, CAE, CAE, CAE, CAE, CAE, CAE, CAE, CAE, CAE, CAE,
                            CAE, CAE, 2 * CAE };
    const int wN[14]    = { CAE, CAE, CAE, CAE, CAE, CAE, CAE, CAE, CAE, CAE, CAE,
                            2 * CAE, 2 * CAE, CAE };
    _Float16* hW[NBLK][14];
    for (int b = 0; b < NBLK; ++b)
        for (int j = 0; j < 14; ++j)
            hW[b][j] = halloc((size_t)wK[j] * wN[j]);
    _Float16* hproj = halloc((size_t)CAE * CTE);

    // ---- weight conversion (once) ----
    for (int b = 0; b < NBLK; ++b)
        for (int j = 0; j < 14; ++j) {
            int kn = wK[j] * wN[j];
            convert_wt<<<kn / 256, 256, 0, stream>>>(BP(b, wsel[j]), hW[b][j], wK[j], wN[j]);
        }
    convert_wt<<<(CAE * CTE) / 256, 256, 0, stream>>>(W_proj, hproj, CAE, CTE);

    init_kernel<<<NATOMS, 128, 0, stream>>>(ref_pos, W_pos, W_cond, outC, sn, aact, cc);
    pair_kernel<<<(NATOMS * NATOMS) / 256, 256, 0, stream>>>(
        ref_pos, uid, cc, W_off, W_inv, W_maskw, W_p1, W_p2, W_p3, outPR);
    bias_kernel<<<(NATOMS * NKW) / 256, 256, 0, stream>>>(
        outPR, BP(0, 0), BP(1, 0), BP(2, 0), biasws);

    dim3 g128(NATOMS / 16, 1), g256(NATOMS / 16, 2), g384(NATOMS / 16, 3);
    for (int b = 0; b < NBLK; ++b) {
        // adaLN gates from sn (fixed across the dataflow)
        wmma_gemm<EP_SIGMOID, true, false><<<g128, 256, 0, stream>>>(
            sn, hW[b][0], BP(b, 2), nullptr, sig_g, nullptr, NATOMS, CAE, CAE);
        wmma_gemm<EP_NONE, false, false><<<g128, 256, 0, stream>>>(
            sn, hW[b][1], nullptr, nullptr, b_ab, nullptr, NATOMS, CAE, CAE);
        wmma_gemm<EP_SIGMOID_M2, false, false><<<g128, 256, 0, stream>>>(
            sn, hW[b][7], nullptr, nullptr, gate_o, nullptr, NATOMS, CAE, CAE);
        wmma_gemm<EP_SIGMOID, true, false><<<g128, 256, 0, stream>>>(
            sn, hW[b][8], BP(b, 12), nullptr, sig_t, nullptr, NATOMS, CAE, CAE);
        wmma_gemm<EP_NONE, false, false><<<g128, 256, 0, stream>>>(
            sn, hW[b][9], nullptr, nullptr, b_tb, nullptr, NATOMS, CAE, CAE);
        wmma_gemm<EP_SIGMOID_M2, false, false><<<g128, 256, 0, stream>>>(
            sn, hW[b][10], nullptr, nullptr, gate_t, nullptr, NATOMS, CAE, CAE);
        // attention
        adaln_kernel<<<NATOMS, 128, 0, stream>>>(aact, sig_g, b_ab, ai);
        wmma_gemm<EP_NONE, true, true><<<g128, 256, 0, stream>>>(
            ai, hW[b][2], BP(b, 5), nullptr, nullptr, qh, NATOMS, CAE, CAE);
        wmma_gemm<EP_NONE, false, true><<<g128, 256, 0, stream>>>(
            ai, hW[b][3], nullptr, nullptr, nullptr, kh, NATOMS, CAE, CAE);
        wmma_gemm<EP_NONE, false, true><<<g128, 256, 0, stream>>>(
            ai, hW[b][4], nullptr, nullptr, nullptr, vh, NATOMS, CAE, CAE);
        wmma_gemm<EP_SIGMOID, false, false><<<g128, 256, 0, stream>>>(
            ai, hW[b][5], nullptr, nullptr, gbuf, nullptr, NATOMS, CAE, CAE);
        attn_kernel<<<dim3(NATOMS / NQW, NHEAD), 128, 0, stream>>>(
            qh, kh, vh, biasws + (size_t)b * NHEAD * NATOMS * NKW, gbuf, obuf);
        wmma_gemm<EP_GATED_ACC, false, false><<<g128, 256, 0, stream>>>(
            obuf, hW[b][6], nullptr, gate_o, aact, nullptr, NATOMS, CAE, CAE);
        // conditioned transition (SwiGLU)
        adaln_kernel<<<NATOMS, 128, 0, stream>>>(aact, sig_t, b_tb, ti);
        wmma_gemm<EP_SILU, false, false><<<g256, 256, 0, stream>>>(
            ti, hW[b][11], nullptr, nullptr, t1, nullptr, NATOMS, CAE, 2 * CAE);
        wmma_gemm<EP_MUL, false, false><<<g256, 256, 0, stream>>>(
            ti, hW[b][12], nullptr, t1, tsw, nullptr, NATOMS, CAE, 2 * CAE);
        wmma_gemm<EP_GATED_ACC, false, false><<<g128, 256, 0, stream>>>(
            tsw, hW[b][13], nullptr, gate_t, aact, nullptr, NATOMS, 2 * CAE, CAE);
    }
    wmma_gemm<EP_RELU, false, false><<<g384, 256, 0, stream>>>(
        aact, hproj, nullptr, nullptr, outQ, nullptr, NATOMS, CAE, CTE);
    pool_kernel<<<(384 * CTE) / 256, 256, 0, stream>>>(outQ, outA);
}
